// SANIModel_21878563406032
// MI455X (gfx1250) — matrix operand order repaired and checked by reading, not compiled, not run on hardware
//
#include <hip/hip_runtime.h>

typedef __attribute__((ext_vector_type(16))) _Float16 v16h;
typedef __attribute__((ext_vector_type(8)))  _Float16 v8h;
typedef __attribute__((ext_vector_type(8)))  float    v8f;

// Problem constants (match reference)
constexpr int T  = 4;
constexpr int D  = 384;
constexpr int H0 = 160, H1 = 128, H2 = 96;
constexpr int BATCH = 2048, A = 64;
constexpr int N = BATCH * A;             // 131072 atoms
constexpr float ALPHA = 0.1f;

// WMMA tiling
constexpr int NT0 = H0 / 16, KT0 = D  / 32;   // 10, 12
constexpr int NT1 = H1 / 16, KT1 = H0 / 32;   //  8,  5
constexpr int NT2 = H2 / 16, KT2 = H1 / 32;   //  6,  4

// LDS activation row stride in halves (336B rows -> lanes 0..15 hit distinct banks)
constexpr int LSTRIDE = 168;

// Workspace layout (bytes)
constexpr size_t OFF_COUNTS = 0;
constexpr size_t OFF_IDX    = 256;
constexpr size_t OFF_E      = OFF_IDX + (size_t)T * N * sizeof(int);       // 2 MB lists
constexpr size_t OFF_W0     = OFF_E   + (size_t)N * sizeof(float);         // 512 KB energies
constexpr size_t SZ_W0      = (size_t)T * NT0 * KT0 * 512 * sizeof(_Float16);
constexpr size_t OFF_W1     = OFF_W0 + SZ_W0;
constexpr size_t SZ_W1      = (size_t)T * NT1 * KT1 * 512 * sizeof(_Float16);
constexpr size_t OFF_W2     = OFF_W1 + SZ_W1;

__device__ __forceinline__ float celu_f(float x) {
    return x > 0.0f ? x : ALPHA * (__expf(x * (1.0f / ALPHA)) - 1.0f);
}

__global__ void zero_counts_kernel(int* counts) {
    if (threadIdx.x < T) counts[threadIdx.x] = 0;
}

// Scatter atoms into per-species index lists.
__global__ void scatter_species_kernel(const int* __restrict__ species,
                                       int* __restrict__ counts,
                                       int* __restrict__ idxbuf) {
    int n = blockIdx.x * blockDim.x + threadIdx.x;
    if (n >= N) return;
    int t = species[n];
    int pos = atomicAdd(&counts[t], 1);
    idxbuf[t * N + pos] = n;
}

// Convert fp32 weights [T, K, H] into WMMA B-fragment-major f16:
// frag f = (t*NT + nt)*KT + kt, lane L holds 16 contiguous halves:
//   L<16 : column n = nt*16+L,      K = kt*32 + (i<8 ? i : i+8)        (i = 0..15)
//   L>=16: column n = nt*16+(L-16), K = kt*32 + 8 + (i<8 ? i : i+8)
__global__ void prep_weights_kernel(const float* __restrict__ src,
                                    _Float16* __restrict__ dst,
                                    int K, int Hc, int NT, int KT) {
    int gid = blockIdx.x * blockDim.x + threadIdx.x;
    int total = T * NT * KT * 32;
    if (gid >= total) return;
    int lane = gid & 31;
    int f    = gid >> 5;
    int kt   = f % KT;
    int nt   = (f / KT) % NT;
    int t    = f / (KT * NT);
    int n    = nt * 16 + (lane & 15);
    int kb   = kt * 32 + ((lane & 16) ? 8 : 0);
    _Float16* out = dst + (size_t)f * 512 + lane * 16;
    const float* s = src + (size_t)t * K * Hc + n;
#pragma unroll
    for (int i = 0; i < 16; ++i)
        out[i] = (_Float16)s[(size_t)(kb + (i < 8 ? i : i + 8)) * Hc];
}

// B fragment load: per-lane base pointer + compile-time element offset, so the
// offset folds into the global_load_b128 immediate.
__device__ __forceinline__ v16h load_bfrag_imm(const _Float16* lanebase, int elem_off) {
    union { v16h h; int4 q[2]; } u;
    const int4* p = (const int4*)(lanebase + elem_off);
    u.q[0] = p[0];
    u.q[1] = p[1];
    return u.h;
}

// A fragment from a row-major fp32 row (ISA 16-bit A layout).
__device__ __forceinline__ v16h load_afrag_f32(const float* arow, int kt, int koff) {
    const float4* p0 = (const float4*)(arow + kt * 32 + koff);
    const float4* p1 = (const float4*)(arow + kt * 32 + 16 + koff);
    float4 x0 = p0[0], x1 = p0[1];
    float4 y0 = p1[0], y1 = p1[1];
    v16h a;
    a[0] = (_Float16)x0.x; a[1] = (_Float16)x0.y; a[2]  = (_Float16)x0.z; a[3]  = (_Float16)x0.w;
    a[4] = (_Float16)x1.x; a[5] = (_Float16)x1.y; a[6]  = (_Float16)x1.z; a[7]  = (_Float16)x1.w;
    a[8] = (_Float16)y0.x; a[9] = (_Float16)y0.y; a[10] = (_Float16)y0.z; a[11] = (_Float16)y0.w;
    a[12] = (_Float16)y1.x; a[13] = (_Float16)y1.y; a[14] = (_Float16)y1.z; a[15] = (_Float16)y1.w;
    return a;
}

// A fragment from an f16 LDS activation row.
__device__ __forceinline__ v16h load_afrag_lds(const _Float16* rowp, int kt, int koff) {
    const v8h* p0 = (const v8h*)(rowp + kt * 32 + koff);
    const v8h* p1 = (const v8h*)(rowp + kt * 32 + 16 + koff);
    v8h lo = *p0, hi = *p1;
    v16h a;
#pragma unroll
    for (int i = 0; i < 8; ++i) { a[i] = lo[i]; a[8 + i] = hi[i]; }
    return a;
}

#define WMMA_F16(af, bf, cf) \
    __builtin_amdgcn_wmma_f32_16x16x32_f16(false, (af), false, (bf), (short)0, (cf), false, false)

// Each wave: 32 atoms (two 16-row tiles sharing every B fragment) of one
// species through the full MLP. blockDim = 128 (4 waves). grid = (N/128, T).
__global__ void __launch_bounds__(128)
mlp_kernel(const float* __restrict__ aev,
           const int* __restrict__ counts,
           const int* __restrict__ idxbuf,
           const _Float16* __restrict__ w0f,
           const _Float16* __restrict__ w1f,
           const _Float16* __restrict__ w2f,
           const float* __restrict__ b0,
           const float* __restrict__ b1,
           const float* __restrict__ b2,
           const float* __restrict__ w3,
           const float* __restrict__ b3,
           float* __restrict__ per_atom) {
    const int s   = blockIdx.y;
    const int cnt = counts[s];
    if ((int)blockIdx.x * 128 >= cnt) return;   // whole block empty: uniform exit

    const int wave = threadIdx.x >> 5;
    const int lane = threadIdx.x & 31;
    const int base  = blockIdx.x * 128 + wave * 32;
    const int tbase = min(base, cnt - 1);                 // clamp empty waves
    const int mcnt  = max(0, min(32, cnt - base));
    const int mmax  = max(mcnt - 1, 0);
    const int* gidx = idxbuf + s * N;

    __shared__ __align__(16) _Float16 act[4][32 * LSTRIDE];
    _Float16* my = act[wave];

    const int col   = lane & 15;
    const int rbase = (lane & 16) ? 8 : 0;
    const int koff  = (lane & 16) ? 8 : 0;
    const int r0 = col;            // this lane's row in tile 0
    const int r1 = 16 + col;       // this lane's row in tile 1
    const int ridx0 = gidx[tbase + min(r0, mmax)];
    const int ridx1 = gidx[tbase + min(r1, mmax)];

    // ---- Load A fragments for layer 0 (fp32 AEV -> f16)
    const float* arow0 = aev + (size_t)ridx0 * D;
    const float* arow1 = aev + (size_t)ridx1 * D;
    v16h af0[KT0], af1[KT0];
#pragma unroll
    for (int kt = 0; kt < KT0; ++kt) {
        af0[kt] = load_afrag_f32(arow0, kt, koff);
        af1[kt] = load_afrag_f32(arow1, kt, koff);
    }

    // ---- Layer 0: [32xD] x [DxH0]
    const _Float16* w0s = w0f + (size_t)s * (NT0 * KT0 * 512) + lane * 16;
    const float*    b0s = b0 + s * H0;
    for (int nt = 0; nt < NT0; ++nt) {
        v8f acc0 = {}, acc1 = {};
#pragma unroll
        for (int kt = 0; kt < KT0; ++kt) {
            v16h b = load_bfrag_imm(w0s, (nt * KT0 + kt) * 512);
            acc0 = WMMA_F16(af0[kt], b, acc0);
            acc1 = WMMA_F16(af1[kt], b, acc1);
        }
        float bias = b0s[nt * 16 + col];
#pragma unroll
        for (int r = 0; r < 8; ++r) {
            my[(rbase + r) * LSTRIDE + nt * 16 + col]        = (_Float16)celu_f(acc0[r] + bias);
            my[(16 + rbase + r) * LSTRIDE + nt * 16 + col]   = (_Float16)celu_f(acc1[r] + bias);
        }
    }
    __syncthreads();

    // ---- Layer 1 A fragments from LDS
    v16h a10[KT1], a11[KT1];
#pragma unroll
    for (int kt = 0; kt < KT1; ++kt) {
        a10[kt] = load_afrag_lds(my + r0 * LSTRIDE, kt, koff);
        a11[kt] = load_afrag_lds(my + r1 * LSTRIDE, kt, koff);
    }
    __syncthreads();

    // ---- Layer 1: [32xH0] x [H0xH1]
    const _Float16* w1s = w1f + (size_t)s * (NT1 * KT1 * 512) + lane * 16;
    const float*    b1s = b1 + s * H1;
    for (int nt = 0; nt < NT1; ++nt) {
        v8f acc0 = {}, acc1 = {};
#pragma unroll
        for (int kt = 0; kt < KT1; ++kt) {
            v16h b = load_bfrag_imm(w1s, (nt * KT1 + kt) * 512);
            acc0 = WMMA_F16(a10[kt], b, acc0);
            acc1 = WMMA_F16(a11[kt], b, acc1);
        }
        float bias = b1s[nt * 16 + col];
#pragma unroll
        for (int r = 0; r < 8; ++r) {
            my[(rbase + r) * LSTRIDE + nt * 16 + col]        = (_Float16)celu_f(acc0[r] + bias);
            my[(16 + rbase + r) * LSTRIDE + nt * 16 + col]   = (_Float16)celu_f(acc1[r] + bias);
        }
    }
    __syncthreads();

    // ---- Layer 2 A fragments from LDS
    v16h a20[KT2], a21[KT2];
#pragma unroll
    for (int kt = 0; kt < KT2; ++kt) {
        a20[kt] = load_afrag_lds(my + r0 * LSTRIDE, kt, koff);
        a21[kt] = load_afrag_lds(my + r1 * LSTRIDE, kt, koff);
    }
    __syncthreads();

    // ---- Layer 2 + fused final dot with W3 (H2 -> 1)
    const _Float16* w2s = w2f + (size_t)s * (NT2 * KT2 * 512) + lane * 16;
    const float*    b2s = b2 + s * H2;
    const float*    w3s = w3 + s * H2;
    float epart0[8] = {}, epart1[8] = {};
    for (int nt = 0; nt < NT2; ++nt) {
        v8f acc0 = {}, acc1 = {};
#pragma unroll
        for (int kt = 0; kt < KT2; ++kt) {
            v16h b = load_bfrag_imm(w2s, (nt * KT2 + kt) * 512);
            acc0 = WMMA_F16(a20[kt], b, acc0);
            acc1 = WMMA_F16(a21[kt], b, acc1);
        }
        float bias = b2s[nt * 16 + col];
        float w3v  = w3s[nt * 16 + col];
#pragma unroll
        for (int r = 0; r < 8; ++r) {
            epart0[r] += celu_f(acc0[r] + bias) * w3v;
            epart1[r] += celu_f(acc1[r] + bias) * w3v;
        }
    }

    // Reduce across the 16-lane halves (columns of the C tiles)
#pragma unroll
    for (int r = 0; r < 8; ++r) {
        float v0 = epart0[r];
        v0 += __shfl_xor(v0, 1);
        v0 += __shfl_xor(v0, 2);
        v0 += __shfl_xor(v0, 4);
        v0 += __shfl_xor(v0, 8);
        epart0[r] = v0;
        float v1 = epart1[r];
        v1 += __shfl_xor(v1, 1);
        v1 += __shfl_xor(v1, 2);
        v1 += __shfl_xor(v1, 4);
        v1 += __shfl_xor(v1, 8);
        epart1[r] = v1;
    }
    if (col == 0) {             // lanes 0 (rows 0..7 / 16..23) and 16 (rows 8..15 / 24..31)
        float b3v = b3[s];
#pragma unroll
        for (int r = 0; r < 8; ++r) {
            int m0 = rbase + r;
            int m1 = 16 + rbase + r;
            if (m0 < mcnt) per_atom[gidx[tbase + m0]] = epart0[r] + b3v;
            if (m1 < mcnt) per_atom[gidx[tbase + m1]] = epart1[r] + b3v;
        }
    }
}

// Deterministic fixed-order per-molecule sum over A=64 atoms.
__global__ void reduce_mol_kernel(const float* __restrict__ per_atom,
                                  float* __restrict__ out) {
    int wave = threadIdx.x >> 5;
    int lane = threadIdx.x & 31;
    int mol = blockIdx.x * (blockDim.x >> 5) + wave;
    if (mol >= BATCH) return;
    const float* e = per_atom + (size_t)mol * A;
    float v = e[lane] + e[lane + 32];
    v += __shfl_xor(v, 16);
    v += __shfl_xor(v, 8);
    v += __shfl_xor(v, 4);
    v += __shfl_xor(v, 2);
    v += __shfl_xor(v, 1);
    if (lane == 0) out[mol] = v;
}

extern "C" void kernel_launch(void* const* d_in, const int* in_sizes, int n_in,
                              void* d_out, int out_size, void* d_ws, size_t ws_size,
                              hipStream_t stream) {
    const int*   species = (const int*)d_in[0];
    const float* aev = (const float*)d_in[1];
    const float* W0 = (const float*)d_in[2];
    const float* b0 = (const float*)d_in[3];
    const float* W1 = (const float*)d_in[4];
    const float* b1 = (const float*)d_in[5];
    const float* W2 = (const float*)d_in[6];
    const float* b2 = (const float*)d_in[7];
    const float* W3 = (const float*)d_in[8];
    const float* b3 = (const float*)d_in[9];

    char* ws = (char*)d_ws;
    int*      counts   = (int*)(ws + OFF_COUNTS);
    int*      idxbuf   = (int*)(ws + OFF_IDX);
    float*    per_atom = (float*)(ws + OFF_E);
    _Float16* w0f      = (_Float16*)(ws + OFF_W0);
    _Float16* w1f      = (_Float16*)(ws + OFF_W1);
    _Float16* w2f      = (_Float16*)(ws + OFF_W2);

    // Output tuple: species (int32, N elements) then energies (f32, BATCH elements)
    hipMemcpyAsync(d_out, d_in[0], (size_t)N * sizeof(int),
                   hipMemcpyDeviceToDevice, stream);
    float* out_e = (float*)d_out + N;

    zero_counts_kernel<<<1, 32, 0, stream>>>(counts);

    {
        int tot = T * NT0 * KT0 * 32;
        prep_weights_kernel<<<(tot + 255) / 256, 256, 0, stream>>>(W0, w0f, D, H0, NT0, KT0);
    }
    {
        int tot = T * NT1 * KT1 * 32;
        prep_weights_kernel<<<(tot + 255) / 256, 256, 0, stream>>>(W1, w1f, H0, H1, NT1, KT1);
    }
    {
        int tot = T * NT2 * KT2 * 32;
        prep_weights_kernel<<<(tot + 255) / 256, 256, 0, stream>>>(W2, w2f, H1, H2, NT2, KT2);
    }

    scatter_species_kernel<<<(N + 255) / 256, 256, 0, stream>>>(species, counts, idxbuf);

    dim3 gmlp((N + 127) / 128, T);
    mlp_kernel<<<gmlp, 128, 0, stream>>>(aev, counts, idxbuf, w0f, w1f, w2f,
                                         b0, b1, b2, W3, b3, per_atom);

    reduce_mol_kernel<<<BATCH / 8, 256, 0, stream>>>(per_atom, out_e);

    (void)in_sizes; (void)n_in; (void)out_size; (void)ws_size;
}
